// HeteroGraphormerLayerComplete_56882546868474
// MI455X (gfx1250) — matrix operand order repaired and checked by reading, not compiled, not run on hardware
//
#include <hip/hip_runtime.h>

#define C_DIM   256
#define H_DIM   8
#define NUM_DST 256
#define MAX_SPD 3
#define MSG_CHUNKS 64

typedef __bf16 bf16_t;
typedef bf16_t v16bf __attribute__((ext_vector_type(16)));
typedef float  v8f   __attribute__((ext_vector_type(8)));

// ---------- helpers ----------
__device__ __forceinline__ unsigned short f2bf(float f) {
  bf16_t b = (bf16_t)f;
  return __builtin_bit_cast(unsigned short, b);
}
__device__ __forceinline__ unsigned packbf(float lo, float hi) {
  return ((unsigned)f2bf(hi) << 16) | (unsigned)f2bf(lo);
}
__device__ __forceinline__ float bflo(unsigned u) { return __builtin_bit_cast(float, u << 16); }
__device__ __forceinline__ float bfhi(unsigned u) { return __builtin_bit_cast(float, u & 0xFFFF0000u); }
// order-preserving float<->uint encoding for atomicMax-based segment max
__device__ __forceinline__ unsigned fenc(float f) {
  unsigned u = __builtin_bit_cast(unsigned, f);
  return (u & 0x80000000u) ? ~u : (u | 0x80000000u);
}
__device__ __forceinline__ float fdec(unsigned k) {
  unsigned u = (k & 0x80000000u) ? (k & 0x7FFFFFFFu) : ~k;
  return __builtin_bit_cast(float, u);
}

// ---------- weight f32 -> bf16 conversion (3 x 256x256) ----------
__global__ __launch_bounds__(256) void cvt_w_kernel(
    const float* __restrict__ Wq, const float* __restrict__ Wk,
    const float* __restrict__ Wv, unsigned short* __restrict__ Wqb,
    unsigned short* __restrict__ Wkb, unsigned short* __restrict__ Wvb) {
  int i = blockIdx.x * 256 + threadIdx.x;
  int m = i >> 16, o = i & 65535;
  const float* s = (m == 0) ? Wq : (m == 1) ? Wk : Wv;
  unsigned short* d = (m == 0) ? Wqb : (m == 1) ? Wkb : Wvb;
  d[o] = f2bf(s[o]);
}

// ---------- workspace init ----------
__global__ __launch_bounds__(256) void init_kernel(
    unsigned* FA, unsigned* FB, int* spd, float* deg,
    unsigned* segmax, float* segsum, int N, int E) {
  int i = blockIdx.x * 256 + threadIdx.x;
  int n8 = N * 8;
  if (i < n8) {
    int node = i >> 3, w = i & 7;
    FA[i] = (node < NUM_DST && (node >> 5) == w) ? (1u << (node & 31)) : 0u;
    FB[i] = 0u;
  }
  if (i < E) spd[i] = MAX_SPD + 1;
  if (i < N) deg[i] = 0.0f;
  if (i < NUM_DST * H_DIM) { segmax[i] = 0u; segsum[i] = 0.0f; }
}

__global__ __launch_bounds__(256) void zero_u32_kernel(unsigned* p, int n) {
  int i = blockIdx.x * 256 + threadIdx.x;
  if (i < n) p[i] = 0u;
}

// ---------- WMMA GEMM (single output, f32 store): used for Q (256 rows) ----------
__global__ __launch_bounds__(256) void wmma_gemm_q_kernel(
    const float* __restrict__ X, const unsigned* __restrict__ Wd,
    const float* __restrict__ bias, float* __restrict__ outF) {
  __shared__ unsigned Axs[16 * 132];
  const int row0 = blockIdx.x * 16;
  const int tid = threadIdx.x;
  {
    int r = tid >> 4;
    int cb = (tid & 15) * 16;
    const float4* xp = (const float4*)(X + (size_t)(row0 + r) * C_DIM + cb);
    float4 f0 = xp[0], f1 = xp[1], f2 = xp[2], f3 = xp[3];
    unsigned* d = &Axs[r * 132 + (cb >> 1)];
    d[0] = packbf(f0.x, f0.y); d[1] = packbf(f0.z, f0.w);
    d[2] = packbf(f1.x, f1.y); d[3] = packbf(f1.z, f1.w);
    d[4] = packbf(f2.x, f2.y); d[5] = packbf(f2.z, f2.w);
    d[6] = packbf(f3.x, f3.y); d[7] = packbf(f3.z, f3.w);
  }
  __syncthreads();

  const int wv = tid >> 5, l = tid & 31;
  const int m = l & 15, khalf = l >> 4;
  const int c0 = wv * 32 + m;
  const int c1 = c0 + 16;
  const unsigned* w0 = Wd + (size_t)c0 * 128;
  const unsigned* w1 = Wd + (size_t)c1 * 128;

  v8f acc0 = {0.f, 0.f, 0.f, 0.f, 0.f, 0.f, 0.f, 0.f};
  v8f acc1 = {0.f, 0.f, 0.f, 0.f, 0.f, 0.f, 0.f, 0.f};
  union { v16bf v; unsigned u[8]; } af, b0, b1;

#pragma unroll
  for (int ks = 0; ks < 8; ++ks) {
#pragma unroll
    for (int j = 0; j < 8; ++j) {
      int k0 = ks * 32 + khalf * 16 + 2 * j;
      af.u[j] = Axs[m * 132 + (k0 >> 1)];
    }
    int base = ks * 16 + khalf * 8;
    uint4 q0 = *(const uint4*)(w0 + base);
    uint4 q1 = *(const uint4*)(w0 + base + 4);
    b0.u[0] = q0.x; b0.u[1] = q0.y; b0.u[2] = q0.z; b0.u[3] = q0.w;
    b0.u[4] = q1.x; b0.u[5] = q1.y; b0.u[6] = q1.z; b0.u[7] = q1.w;
    uint4 p0 = *(const uint4*)(w1 + base);
    uint4 p1 = *(const uint4*)(w1 + base + 4);
    b1.u[0] = p0.x; b1.u[1] = p0.y; b1.u[2] = p0.z; b1.u[3] = p0.w;
    b1.u[4] = p1.x; b1.u[5] = p1.y; b1.u[6] = p1.z; b1.u[7] = p1.w;

    acc0 = __builtin_amdgcn_wmma_f32_16x16x32_bf16(false, af.v, false, b0.v,
                                                   (short)0, acc0, false, false);
    acc1 = __builtin_amdgcn_wmma_f32_16x16x32_bf16(false, af.v, false, b1.v,
                                                   (short)0, acc1, false, false);
  }

  union { v8f v; float f[8]; } a0, a1;
  a0.v = acc0; a1.v = acc1;
  float bia0 = bias[c0], bia1 = bias[c1];
#pragma unroll
  for (int r = 0; r < 8; ++r) {
    int mr = r + khalf * 8;
    outF[(size_t)(row0 + mr) * C_DIM + c0] = a0.f[r] + bia0;
    outF[(size_t)(row0 + mr) * C_DIM + c1] = a1.f[r] + bia1;
  }
}

// ---------- fused K+V WMMA GEMM with async global->LDS A staging ----------
// Block: 8 waves, 16 rows x 256 cols for BOTH K and V (x read once).
// A tile staged as f32 via GLOBAL_LOAD_ASYNC_TO_LDS_B128 (ASYNCcnt path),
// converted to bf16 fragments on read (pack co-executes with WMMA).
__global__ __launch_bounds__(256) void wmma_gemm_kv_kernel(
    const float* __restrict__ X,
    const unsigned* __restrict__ Wkd, const unsigned* __restrict__ Wvd,
    const float* __restrict__ bk, const float* __restrict__ bv,
    unsigned short* __restrict__ Kb, unsigned short* __restrict__ Vb) {
  __shared__ float Axs[16 * 260];   // f32 tile, padded row stride 260 dwords
  const int row0 = blockIdx.x * 16;
  const int tid = threadIdx.x;

  // Async stage: each thread copies 64B (4 x b128) of its row chunk.
  {
    int r = tid >> 4, q = tid & 15;
    const float* g = X + (size_t)(row0 + r) * C_DIM + q * 16;
    unsigned lds = (unsigned)((r * 260 + q * 16) * 4);   // LDS byte address
#pragma unroll
    for (int i = 0; i < 4; ++i) {
      asm volatile("global_load_async_to_lds_b128 %0, %1, off"
                   :: "v"(lds + i * 16), "v"(g + i * 4) : "memory");
    }
    asm volatile("s_wait_asynccnt 0x0" ::: "memory");
  }
  __syncthreads();

  const int wv = tid >> 5, l = tid & 31;
  const int m = l & 15, khalf = l >> 4;
  const int c0 = wv * 32 + m;
  const int c1 = c0 + 16;
  const unsigned* wk0 = Wkd + (size_t)c0 * 128;
  const unsigned* wk1 = Wkd + (size_t)c1 * 128;
  const unsigned* wv0 = Wvd + (size_t)c0 * 128;
  const unsigned* wv1 = Wvd + (size_t)c1 * 128;

  v8f ak0 = {0.f,0.f,0.f,0.f,0.f,0.f,0.f,0.f};
  v8f ak1 = {0.f,0.f,0.f,0.f,0.f,0.f,0.f,0.f};
  v8f av0 = {0.f,0.f,0.f,0.f,0.f,0.f,0.f,0.f};
  v8f av1 = {0.f,0.f,0.f,0.f,0.f,0.f,0.f,0.f};
  union { v16bf v; unsigned u[8]; } af, bk0f, bk1f, bv0f, bv1f;

#pragma unroll
  for (int ks = 0; ks < 8; ++ks) {
    // A fragment from f32 LDS, packed to bf16 on read
#pragma unroll
    for (int j = 0; j < 8; ++j) {
      int k0 = ks * 32 + khalf * 16 + 2 * j;
      float2 xv = *(const float2*)(&Axs[m * 260 + k0]);
      af.u[j] = packbf(xv.x, xv.y);
    }
    int base = ks * 16 + khalf * 8;
    uint4 t0, t1;
    t0 = *(const uint4*)(wk0 + base); t1 = *(const uint4*)(wk0 + base + 4);
    bk0f.u[0]=t0.x; bk0f.u[1]=t0.y; bk0f.u[2]=t0.z; bk0f.u[3]=t0.w;
    bk0f.u[4]=t1.x; bk0f.u[5]=t1.y; bk0f.u[6]=t1.z; bk0f.u[7]=t1.w;
    t0 = *(const uint4*)(wk1 + base); t1 = *(const uint4*)(wk1 + base + 4);
    bk1f.u[0]=t0.x; bk1f.u[1]=t0.y; bk1f.u[2]=t0.z; bk1f.u[3]=t0.w;
    bk1f.u[4]=t1.x; bk1f.u[5]=t1.y; bk1f.u[6]=t1.z; bk1f.u[7]=t1.w;
    t0 = *(const uint4*)(wv0 + base); t1 = *(const uint4*)(wv0 + base + 4);
    bv0f.u[0]=t0.x; bv0f.u[1]=t0.y; bv0f.u[2]=t0.z; bv0f.u[3]=t0.w;
    bv0f.u[4]=t1.x; bv0f.u[5]=t1.y; bv0f.u[6]=t1.z; bv0f.u[7]=t1.w;
    t0 = *(const uint4*)(wv1 + base); t1 = *(const uint4*)(wv1 + base + 4);
    bv1f.u[0]=t0.x; bv1f.u[1]=t0.y; bv1f.u[2]=t0.z; bv1f.u[3]=t0.w;
    bv1f.u[4]=t1.x; bv1f.u[5]=t1.y; bv1f.u[6]=t1.z; bv1f.u[7]=t1.w;

    ak0 = __builtin_amdgcn_wmma_f32_16x16x32_bf16(false, af.v, false, bk0f.v,
                                                  (short)0, ak0, false, false);
    ak1 = __builtin_amdgcn_wmma_f32_16x16x32_bf16(false, af.v, false, bk1f.v,
                                                  (short)0, ak1, false, false);
    av0 = __builtin_amdgcn_wmma_f32_16x16x32_bf16(false, af.v, false, bv0f.v,
                                                  (short)0, av0, false, false);
    av1 = __builtin_amdgcn_wmma_f32_16x16x32_bf16(false, af.v, false, bv1f.v,
                                                  (short)0, av1, false, false);
  }

  union { v8f v; float f[8]; } k0u, k1u, v0u, v1u;
  k0u.v = ak0; k1u.v = ak1; v0u.v = av0; v1u.v = av1;
  float bk0s = bk[c0], bk1s = bk[c1], bv0s = bv[c0], bv1s = bv[c1];
#pragma unroll
  for (int r = 0; r < 8; ++r) {
    int mr = r + khalf * 8;  // C/D layout: M = r + 8*(lane>>4), N = lane&15
    size_t ro = (size_t)(row0 + mr) * C_DIM;
    Kb[ro + c0] = f2bf(k0u.f[r] + bk0s);
    Kb[ro + c1] = f2bf(k1u.f[r] + bk1s);
    Vb[ro + c0] = f2bf(v0u.f[r] + bv0s);
    Vb[ro + c1] = f2bf(v1u.f[r] + bv1s);
  }
}

// ---------- degree ----------
__global__ __launch_bounds__(256) void deg_kernel(const int* __restrict__ src,
                                                  const int* __restrict__ dst,
                                                  float* deg, int E) {
  int e = blockIdx.x * 256 + threadIdx.x;
  if (e >= E) return;
  atomicAdd(&deg[src[e]], 1.0f);
  atomicAdd(&deg[dst[e]], 1.0f);
}

// ---------- bitmask reverse-BFS: Fnew[dst] |= F[src] ----------
__global__ __launch_bounds__(256) void bfs_propagate_kernel(
    const int* __restrict__ src, const int* __restrict__ dst,
    const unsigned* __restrict__ F, unsigned* __restrict__ Fn, int E) {
  int e = blockIdx.x * 256 + threadIdx.x;
  if (e >= E) return;
  int s = src[e], d = dst[e];
  const uint4* fp = (const uint4*)(F + (size_t)s * 8);
  uint4 a = fp[0], b = fp[1];
  unsigned* o = Fn + (size_t)d * 8;
  if (a.x) atomicOr(o + 0, a.x);
  if (a.y) atomicOr(o + 1, a.y);
  if (a.z) atomicOr(o + 2, a.z);
  if (a.w) atomicOr(o + 3, a.w);
  if (b.x) atomicOr(o + 4, b.x);
  if (b.y) atomicOr(o + 5, b.y);
  if (b.z) atomicOr(o + 6, b.z);
  if (b.w) atomicOr(o + 7, b.w);
}

__global__ __launch_bounds__(256) void spd_update_kernel(
    const int* __restrict__ src, const int* __restrict__ dst,
    const unsigned* __restrict__ F, int* spd, int k, int E) {
  int e = blockIdx.x * 256 + threadIdx.x;
  if (e >= E) return;
  if (spd[e] != MAX_SPD + 1) return;
  int s = src[e], d = dst[e];
  unsigned w = F[(size_t)s * 8 + (d >> 5)];
  if ((w >> (d & 31)) & 1u) spd[e] = k;
}

// ---------- per-edge scores: wave per edge, 4 lanes per head ----------
__global__ __launch_bounds__(256) void scores_kernel(
    const int* __restrict__ src, const int* __restrict__ dst,
    const float* __restrict__ Qf, const unsigned short* __restrict__ Kb,
    const int* __restrict__ spd, const float* __restrict__ spdw,
    float* __restrict__ sc, unsigned* __restrict__ segmax, int E) {
  int wv = threadIdx.x >> 5, l = threadIdx.x & 31;
  int e = blockIdx.x * 8 + wv;
  if (e >= E) return;
  int s = src[e], d = dst[e];
  int c = l * 8;
  const float4* qp = (const float4*)(Qf + (size_t)d * C_DIM + c);
  float4 q0 = qp[0], q1 = qp[1];
  uint4 kk = *(const uint4*)(Kb + (size_t)s * C_DIM + c);
  float p = q0.x * bflo(kk.x) + q0.y * bfhi(kk.x)
          + q0.z * bflo(kk.y) + q0.w * bfhi(kk.y)
          + q1.x * bflo(kk.z) + q1.y * bfhi(kk.z)
          + q1.z * bflo(kk.w) + q1.w * bfhi(kk.w);
  p += __shfl_xor(p, 1, 32);
  p += __shfl_xor(p, 2, 32);
  if ((l & 3) == 0) {
    int h = l >> 2;
    float v = p * 0.17677669529663687f + spdw[spd[e] * H_DIM + h];  // 1/sqrt(32)
    sc[(size_t)e * H_DIM + h] = v;
    atomicMax(&segmax[d * H_DIM + h], fenc(v));
  }
}

// ---------- exp(s - max), accumulate segment sums ----------
__global__ __launch_bounds__(256) void exp_kernel(
    const int* __restrict__ dst, float* __restrict__ sc,
    const unsigned* __restrict__ segmax, float* __restrict__ segsum, int EH) {
  int i = blockIdx.x * 256 + threadIdx.x;
  if (i >= EH) return;
  int e = i >> 3, h = i & 7;
  int d = dst[e];
  float mx = fdec(segmax[d * H_DIM + h]);
  float ex = __expf(sc[i] - mx);
  sc[i] = ex;
  atomicAdd(&segsum[d * H_DIM + h], ex);
}

// ---------- weighted message scatter via LDS (ds_add_f32), no global atomics ----
// grid = (MSG_CHUNKS edge chunks, 4 channel quarters); block owns a private
// 256-dst x 64-channel f32 accumulator in LDS (64KB), flushed with plain stores.
__global__ __launch_bounds__(256) void msg_kernel(
    const int* __restrict__ src, const int* __restrict__ dst,
    const unsigned short* __restrict__ Vb, const float* __restrict__ sc,
    const float* __restrict__ segsum, float* __restrict__ part,
    int E, int echunk) {
  __shared__ float lagg[NUM_DST * 64];
  const int tid = threadIdx.x;
  const int quarter = blockIdx.y;
  const int c0 = quarter * 64;
  for (int i = tid; i < NUM_DST * 64; i += 256) lagg[i] = 0.f;
  __syncthreads();

  const int wv = tid >> 5, l = tid & 31;
  const int e0 = blockIdx.x * echunk;
  const int e1 = (e0 + echunk < E) ? (e0 + echunk) : E;
  const int cl = c0 + l * 2;        // 2 channels per lane
  const int h = cl >> 5;            // head for this lane
  for (int e = e0 + wv; e < e1; e += 8) {
    int s = src[e], d = dst[e];
    float w = sc[(size_t)e * H_DIM + h] / segsum[d * H_DIM + h];
    unsigned vv = *(const unsigned*)(Vb + (size_t)s * C_DIM + cl);
    atomicAdd(&lagg[d * 64 + l * 2 + 0], bflo(vv) * w);
    atomicAdd(&lagg[d * 64 + l * 2 + 1], bfhi(vv) * w);
  }
  __syncthreads();
  float* o = part + ((size_t)blockIdx.x * 4 + quarter) * (NUM_DST * 64);
  for (int i = tid; i < NUM_DST * 64; i += 256) o[i] = lagg[i];
}

// ---------- reduce partial aggregates: agg[d][c] = sum over chunks ----------
__global__ __launch_bounds__(256) void msg_reduce_kernel(
    const float* __restrict__ part, float* __restrict__ agg, int chunks) {
  int i = blockIdx.x * 256 + threadIdx.x;   // i = d*256 + c over 65536
  int d = i >> 8, c = i & 255;
  int q = c >> 6, cq = c & 63;
  float s = 0.f;
  for (int ch = 0; ch < chunks; ++ch)
    s += part[((size_t)ch * 4 + q) * (NUM_DST * 64) + d * 64 + cq];
  agg[i] = s;
}

// ---------- residual + degree + LayerNorm: wave per row ----------
__global__ __launch_bounds__(256) void ln_kernel(
    const float* __restrict__ x, const float* __restrict__ agg,
    const float* __restrict__ deg, const float* __restrict__ gamma,
    const float* __restrict__ beta, float* __restrict__ out, int N) {
  int wv = threadIdx.x >> 5, l = threadIdx.x & 31;
  int row = blockIdx.x * 8 + wv;
  if (row >= N) return;
  int c = l * 8;
  const float4* xp = (const float4*)(x + (size_t)row * C_DIM + c);
  float4 x0 = xp[0], x1 = xp[1];
  float v[8] = {x0.x, x0.y, x0.z, x0.w, x1.x, x1.y, x1.z, x1.w};
  if (row < NUM_DST) {
    const float4* ap = (const float4*)(agg + (size_t)row * C_DIM + c);
    float4 a0 = ap[0], a1 = ap[1];
    v[0] += a0.x; v[1] += a0.y; v[2] += a0.z; v[3] += a0.w;
    v[4] += a1.x; v[5] += a1.y; v[6] += a1.z; v[7] += a1.w;
  }
  float dg = deg[row];
  float sum = 0.f, sq = 0.f;
#pragma unroll
  for (int j = 0; j < 8; ++j) { v[j] += dg; sum += v[j]; sq += v[j] * v[j]; }
  for (int off = 16; off; off >>= 1) {
    sum += __shfl_xor(sum, off, 32);
    sq  += __shfl_xor(sq,  off, 32);
  }
  float mu = sum * (1.0f / 256.0f);
  float var = sq * (1.0f / 256.0f) - mu * mu;
  float rstd = rsqrtf(var + 1e-5f);
  float4 o0, o1;
  o0.x = (v[0] - mu) * rstd * gamma[c + 0] + beta[c + 0];
  o0.y = (v[1] - mu) * rstd * gamma[c + 1] + beta[c + 1];
  o0.z = (v[2] - mu) * rstd * gamma[c + 2] + beta[c + 2];
  o0.w = (v[3] - mu) * rstd * gamma[c + 3] + beta[c + 3];
  o1.x = (v[4] - mu) * rstd * gamma[c + 4] + beta[c + 4];
  o1.y = (v[5] - mu) * rstd * gamma[c + 5] + beta[c + 5];
  o1.z = (v[6] - mu) * rstd * gamma[c + 6] + beta[c + 6];
  o1.w = (v[7] - mu) * rstd * gamma[c + 7] + beta[c + 7];
  float4* op = (float4*)(out + (size_t)row * C_DIM + c);
  op[0] = o0; op[1] = o1;
}

// ---------- host launch ----------
extern "C" void kernel_launch(void* const* d_in, const int* in_sizes, int n_in,
                              void* d_out, int out_size, void* d_ws, size_t ws_size,
                              hipStream_t stream) {
  (void)n_in; (void)out_size; (void)ws_size;
  const float* x    = (const float*)d_in[0];
  const int*   src  = (const int*)d_in[1];
  const int*   dst  = (const int*)d_in[2];
  const float* Wq   = (const float*)d_in[3];
  const float* bq   = (const float*)d_in[4];
  const float* Wk   = (const float*)d_in[5];
  const float* bk   = (const float*)d_in[6];
  const float* Wv   = (const float*)d_in[7];
  const float* bv   = (const float*)d_in[8];
  const float* spdw = (const float*)d_in[9];
  const float* gam  = (const float*)d_in[10];
  const float* bet  = (const float*)d_in[11];
  const int N = in_sizes[0] / C_DIM;   // 50000 (multiple of 16)
  const int E = in_sizes[1];           // 300000
  float* out = (float*)d_out;

  char* p = (char*)d_ws;
  auto carve = [&](size_t bytes) -> char* {
    char* r = p;
    p += (bytes + 255) & ~(size_t)255;
    return r;
  };
  unsigned short* Wqb = (unsigned short*)carve((size_t)C_DIM * C_DIM * 2);
  unsigned short* Wkb = (unsigned short*)carve((size_t)C_DIM * C_DIM * 2);
  unsigned short* Wvb = (unsigned short*)carve((size_t)C_DIM * C_DIM * 2);
  float*          Qf  = (float*)carve((size_t)NUM_DST * C_DIM * 4);
  unsigned short* Kb  = (unsigned short*)carve((size_t)N * C_DIM * 2);
  unsigned short* Vb  = (unsigned short*)carve((size_t)N * C_DIM * 2);
  unsigned*       FA  = (unsigned*)carve((size_t)N * 8 * 4);
  unsigned*       FB  = (unsigned*)carve((size_t)N * 8 * 4);
  int*            spd = (int*)carve((size_t)E * 4);
  float*          sc  = (float*)carve((size_t)E * H_DIM * 4);
  unsigned*   segmax  = (unsigned*)carve((size_t)NUM_DST * H_DIM * 4);
  float*      segsum  = (float*)carve((size_t)NUM_DST * H_DIM * 4);
  float*         agg  = (float*)carve((size_t)NUM_DST * C_DIM * 4);
  float*         deg  = (float*)carve((size_t)N * 4);
  float*        part  = (float*)carve((size_t)MSG_CHUNKS * 4 * NUM_DST * 64 * 4);

  const int eb  = (E + 255) / 256;
  const int ewb = (E + 7) / 8;
  const int n8b = (N * 8 + 255) / 256;

  cvt_w_kernel<<<(3 * C_DIM * C_DIM) / 256, 256, 0, stream>>>(Wq, Wk, Wv, Wqb, Wkb, Wvb);
  init_kernel<<<n8b, 256, 0, stream>>>(FA, FB, spd, deg, segmax, segsum, N, E);

  // GEMMs: Q only needs the 256 anchor rows; K+V fused (x read once), bf16 out.
  wmma_gemm_q_kernel<<<NUM_DST / 16, 256, 0, stream>>>(x, (const unsigned*)Wqb, bq, Qf);
  wmma_gemm_kv_kernel<<<N / 16, 256, 0, stream>>>(x, (const unsigned*)Wkb, (const unsigned*)Wvb,
                                                  bk, bv, Kb, Vb);

  deg_kernel<<<eb, 256, 0, stream>>>(src, dst, deg, E);

  // reverse BFS, 3 steps, ping-pong bitmask frontiers
  bfs_propagate_kernel<<<eb, 256, 0, stream>>>(src, dst, FA, FB, E);
  spd_update_kernel<<<eb, 256, 0, stream>>>(src, dst, FB, spd, 1, E);
  zero_u32_kernel<<<n8b, 256, 0, stream>>>(FA, N * 8);
  bfs_propagate_kernel<<<eb, 256, 0, stream>>>(src, dst, FB, FA, E);
  spd_update_kernel<<<eb, 256, 0, stream>>>(src, dst, FA, spd, 2, E);
  zero_u32_kernel<<<n8b, 256, 0, stream>>>(FB, N * 8);
  bfs_propagate_kernel<<<eb, 256, 0, stream>>>(src, dst, FA, FB, E);
  spd_update_kernel<<<eb, 256, 0, stream>>>(src, dst, FB, spd, 3, E);

  scores_kernel<<<ewb, 256, 0, stream>>>(src, dst, Qf, Kb, spd, spdw, sc, segmax, E);
  exp_kernel<<<(E * H_DIM + 255) / 256, 256, 0, stream>>>(dst, sc, segmax, segsum, E * H_DIM);

  const int echunk = (E + MSG_CHUNKS - 1) / MSG_CHUNKS;
  dim3 mg(MSG_CHUNKS, 4);
  msg_kernel<<<mg, 256, 0, stream>>>(src, dst, Vb, sc, segsum, part, E, echunk);
  msg_reduce_kernel<<<(NUM_DST * C_DIM) / 256, 256, 0, stream>>>(part, agg, MSG_CHUNKS);

  ln_kernel<<<(N + 7) / 8, 256, 0, stream>>>(x, agg, deg, gam, bet, out, N);
}
